// SelfAttention_4071628996937
// MI455X (gfx1250) — matrix operand order repaired and checked by reading, not compiled, not run on hardware
//
#include <hip/hip_runtime.h>
#include <hip/hip_bf16.h>
#include <cstdint>

typedef __attribute__((ext_vector_type(16))) __bf16 v16bf;
typedef __attribute__((ext_vector_type(8)))  float  v8f;

#define BM 128
#define BN 128
#define BK 32
#define LDSP 40         // padded LDS row stride (elements); 80B rows keep 16B alignment
#define NTHREADS 256    // 8 waves (wave32)

struct alignas(16) U128 { unsigned int a, b, c, d; };

__device__ __forceinline__ unsigned short f32_to_bf16_rne(float f) {
    unsigned int u = __float_as_uint(f);
    unsigned int r = u + 0x7FFFu + ((u >> 16) & 1u);
    return (unsigned short)(r >> 16);
}

// ---- CDNA5 async global->LDS copy (ASYNCcnt-tracked, no VGPR round trip) ----
__device__ __forceinline__ void async_copy_b128(unsigned lds_byte_addr,
                                                const unsigned short* gptr) {
    asm volatile("global_load_async_to_lds_b128 %0, %1, off"
                 :: "v"(lds_byte_addr), "v"(gptr) : "memory");
}
__device__ __forceinline__ void wait_asynccnt0() {
    asm volatile("s_wait_asynccnt 0x0" ::: "memory");
}

// ---------------- fragment loads from LDS (ISA 7.12.2 layouts) ----------------
// A 16x32 bf16: lanes 0-15 -> rows 0-15, K=0..7 & 16..23; lanes 16-31 -> K=8..15 & 24..31
__device__ __forceinline__ v16bf load_frag_a(const unsigned short* base, int mbase) {
    int l   = threadIdx.x & 31;
    int row = l & 15;
    int kb  = (l < 16) ? 0 : 8;
    const unsigned short* p = base + (mbase + row) * LDSP + kb;
    union UU { U128 q[2]; v16bf v; } u;
    u.q[0] = *(const U128*)(p);
    u.q[1] = *(const U128*)(p + 16);
    return u.v;
}
// B 32x16 bf16 staged as Blds[col][k]: lanes 0-15 -> K=0..7 & 16..23; lanes 16-31 -> K=8..15 & 24..31
__device__ __forceinline__ v16bf load_frag_b(const unsigned short* base, int nbase) {
    int l   = threadIdx.x & 31;
    int col = l & 15;
    int kb  = (l < 16) ? 0 : 8;
    const unsigned short* p = base + (nbase + col) * LDSP + kb;
    union UU { U128 q[2]; v16bf v; } u;
    u.q[0] = *(const U128*)(p);
    u.q[1] = *(const U128*)(p + 16);
    return u.v;
}

// 8 WMMAs on one staged tile pair (wave tile 32x64 = 2x4 16x16 tiles)
__device__ __forceinline__ void compute_tile(const unsigned short* Ab, const unsigned short* Bb,
                                             int wm, int wn, v8f acc[2][4]) {
    v16bf a0 = load_frag_a(Ab, wm);
    v16bf a1 = load_frag_a(Ab, wm + 16);
    v16bf b0 = load_frag_b(Bb, wn);
    v16bf b1 = load_frag_b(Bb, wn + 16);
    v16bf b2 = load_frag_b(Bb, wn + 32);
    v16bf b3 = load_frag_b(Bb, wn + 48);
    acc[0][0] = __builtin_amdgcn_wmma_f32_16x16x32_bf16(false, a0, false, b0, (short)0, acc[0][0], false, false);
    acc[0][1] = __builtin_amdgcn_wmma_f32_16x16x32_bf16(false, a0, false, b1, (short)0, acc[0][1], false, false);
    acc[0][2] = __builtin_amdgcn_wmma_f32_16x16x32_bf16(false, a0, false, b2, (short)0, acc[0][2], false, false);
    acc[0][3] = __builtin_amdgcn_wmma_f32_16x16x32_bf16(false, a0, false, b3, (short)0, acc[0][3], false, false);
    acc[1][0] = __builtin_amdgcn_wmma_f32_16x16x32_bf16(false, a1, false, b0, (short)0, acc[1][0], false, false);
    acc[1][1] = __builtin_amdgcn_wmma_f32_16x16x32_bf16(false, a1, false, b1, (short)0, acc[1][1], false, false);
    acc[1][2] = __builtin_amdgcn_wmma_f32_16x16x32_bf16(false, a1, false, b2, (short)0, acc[1][2], false, false);
    acc[1][3] = __builtin_amdgcn_wmma_f32_16x16x32_bf16(false, a1, false, b3, (short)0, acc[1][3], false, false);
}

// ---------------- generic bf16 WMMA GEMM, D = A*B ----------------
// Operands row-major in their staged sense: A[m][k] (lda), B[n][k] (ldb).
// EPI 0: += bias[m]; rows [0,512) -> transposed bf16 Q/K store; rows [512,1536) -> natural V
// EPI 1: store f32 (scores, already transposed by construction)
// EPI 2: store sigmoid(acc)*x f32 (final output)
template<int EPI>
__global__ __launch_bounds__(NTHREADS, 1)
void gemm_wmma_kernel(const unsigned short* __restrict__ A, long long sA, int lda,
                      const unsigned short* __restrict__ Bm, long long sB, int ldb,
                      void* __restrict__ Out, void* __restrict__ Out2,
                      long long sO, int ldo,
                      const float* __restrict__ Extra, long long sX,
                      int K)
{
    __shared__ __align__(16) unsigned short Al[2][BM * LDSP];
    __shared__ __align__(16) unsigned short Bl[2][BN * LDSP];

    const int bz = blockIdx.z;
    const int m0 = blockIdx.y * BM;
    const int n0 = blockIdx.x * BN;

    const int t = threadIdx.x;
    // staging geometry: 512 16B chunks per operand; thread t -> rows rS, rS+64, fixed chunk col
    const int rS = t >> 2;
    const int cS = (t & 3) * 8;
    const unsigned short* gA = A  + (size_t)bz * sA + (size_t)(m0 + rS) * lda + cS;
    const unsigned short* gB = Bm + (size_t)bz * sB + (size_t)(n0 + rS) * ldb + cS;
    const size_t gA1 = (size_t)64 * lda;
    const size_t gB1 = (size_t)64 * ldb;

    // LDS byte addresses for this thread's 4 chunk destinations (buffer 0)
    const unsigned laA = (unsigned)(size_t)(&Al[0][0] + rS * LDSP + cS);
    const unsigned laB = (unsigned)(size_t)(&Bl[0][0] + rS * LDSP + cS);
    const unsigned ROWSTEP = 64 * LDSP * 2;          // +64 rows, bytes
    const unsigned ABUF    = BM * LDSP * 2;          // buffer 1 offset, bytes
    const unsigned BBUF    = BN * LDSP * 2;

    const int wave = t >> 5;
    const int wm = (wave & 3) * 32;   // 4 waves tile M (32 rows each)
    const int wn = (wave >> 2) * 64;  // 2 waves tile N (64 cols each)

    v8f zero = {};
    v8f acc[2][4];
    #pragma unroll
    for (int i = 0; i < 2; ++i)
        #pragma unroll
        for (int j = 0; j < 4; ++j) acc[i][j] = zero;

    const int nk = K / BK;            // even for all call sites (32, 8, 32)

    // prime buffer 0 with tile 0
    async_copy_b128(laA,           gA);
    async_copy_b128(laA + ROWSTEP, gA + gA1);
    async_copy_b128(laB,           gB);
    async_copy_b128(laB + ROWSTEP, gB + gB1);

    for (int kt = 0; kt < nk; kt += 2) {
        // ---- tile kt (buffer 0) ----
        wait_asynccnt0();
        __syncthreads();
        {   // issue tile kt+1 into buffer 1 (nk even -> kt+1 < nk), overlaps compute
            const int ko = (kt + 1) * BK;
            async_copy_b128(laA + ABUF,           gA + ko);
            async_copy_b128(laA + ABUF + ROWSTEP, gA + gA1 + ko);
            async_copy_b128(laB + BBUF,           gB + ko);
            async_copy_b128(laB + BBUF + ROWSTEP, gB + gB1 + ko);
        }
        if (kt + 4 < nk) {
            __builtin_prefetch(gA + (kt + 4) * BK, 0, 1);
            __builtin_prefetch(gB + (kt + 4) * BK, 0, 1);
        }
        compute_tile(&Al[0][0], &Bl[0][0], wm, wn, acc);

        // ---- tile kt+1 (buffer 1) ----
        wait_asynccnt0();
        __syncthreads();
        if (kt + 2 < nk) {
            const int ko = (kt + 2) * BK;
            async_copy_b128(laA,           gA + ko);
            async_copy_b128(laA + ROWSTEP, gA + gA1 + ko);
            async_copy_b128(laB,           gB + ko);
            async_copy_b128(laB + ROWSTEP, gB + gB1 + ko);
        }
        compute_tile(&Al[1][0], &Bl[1][0], wm, wn, acc);
    }

    // Epilogue. C/D layout: VGPR r, lane l -> M = r + 8*(l>=16), N = l&15
    const int l    = t & 31;
    const int col  = l & 15;
    const int rofs = (l < 16) ? 0 : 8;
    #pragma unroll
    for (int i = 0; i < 2; ++i) {
        const int gmb = m0 + wm + i * 16 + rofs;   // 8 consecutive output rows
        #pragma unroll
        for (int j = 0; j < 4; ++j) {
            const int gn = n0 + wn + j * 16 + col;
            if constexpr (EPI == 0) {
                if (gmb < 512) {
                    // Q/K: store transposed -> [b][sel][n=gn][q=mq..mq+7], 16B packed store
                    const int sel = gmb >> 8, mq = gmb & 255;
                    union PK { unsigned short s[8]; U128 u; } pk;
                    #pragma unroll
                    for (int r = 0; r < 8; ++r)
                        pk.s[r] = f32_to_bf16_rne(acc[i][j][r] + Extra[gmb + r]);
                    *(U128*)((unsigned short*)Out +
                             (size_t)bz * 524288 + (size_t)sel * 262144 +
                             (size_t)gn * 256 + mq) = pk.u;
                } else {
                    // V: natural [b][c][n]
                    #pragma unroll
                    for (int r = 0; r < 8; ++r)
                        ((unsigned short*)Out2)[(size_t)bz * 1048576 +
                                                (size_t)(gmb + r - 512) * 1024 + gn] =
                            f32_to_bf16_rne(acc[i][j][r] + Extra[gmb + r]);
                }
            } else if constexpr (EPI == 1) {
                #pragma unroll
                for (int r = 0; r < 8; ++r)
                    ((float*)Out)[(size_t)bz * sO + (size_t)(gmb + r) * ldo + gn] = acc[i][j][r];
            } else {
                #pragma unroll
                for (int r = 0; r < 8; ++r) {
                    const float xv = Extra[(size_t)bz * sX + (size_t)(gmb + r) * ldo + gn];
                    const float s  = 1.0f / (1.0f + __expf(-acc[i][j][r]));
                    ((float*)Out)[(size_t)bz * sO + (size_t)(gmb + r) * ldo + gn] = s * xv;
                }
            }
        }
    }
}

// ---------------- x[b][c][n] f32 -> xT[b][n][c] bf16, tiled transpose ----------------
__global__ __launch_bounds__(256)
void transpose_cvt_kernel(const float* __restrict__ x, unsigned short* __restrict__ xT,
                          int C, int N)
{
    __shared__ float tile[32][33];
    const int bz = blockIdx.z;
    const int n0 = blockIdx.x * 32, c0 = blockIdx.y * 32;
    const int tx = threadIdx.x & 31, ty = threadIdx.x >> 5;
    const float* xb = x + (size_t)bz * C * N;
    unsigned short* xTb = xT + (size_t)bz * C * N;
    #pragma unroll
    for (int i = 0; i < 4; ++i) {
        const int c = ty + i * 8;
        tile[c][tx] = xb[(size_t)(c0 + c) * N + n0 + tx];
    }
    __syncthreads();
    #pragma unroll
    for (int i = 0; i < 4; ++i) {
        const int n = ty + i * 8;
        xTb[(size_t)(n0 + n) * C + c0 + tx] = f32_to_bf16_rne(tile[tx][n]);
    }
}

// ---------------- f32 -> bf16 linear conversion (weights) ----------------
__global__ __launch_bounds__(256)
void cvt_f32_bf16_kernel(const float* __restrict__ src, unsigned short* __restrict__ dst,
                         long long n4)
{
    long long i      = (long long)blockIdx.x * 256 + threadIdx.x;
    long long stride = (long long)gridDim.x * 256;
    struct F128 { float x, y, z, w; };
    for (; i < n4; i += stride) {
        F128 f = ((const F128*)src)[i];
        unsigned long long packed =
              (unsigned long long)f32_to_bf16_rne(f.x)
            | ((unsigned long long)f32_to_bf16_rne(f.y) << 16)
            | ((unsigned long long)f32_to_bf16_rne(f.z) << 32)
            | ((unsigned long long)f32_to_bf16_rne(f.w) << 48);
        ((unsigned long long*)dst)[i] = packed;
    }
}

__global__ __launch_bounds__(256)
void concat_bias_kernel(const float* __restrict__ bq, const float* __restrict__ bk,
                        const float* __restrict__ bv, float* __restrict__ bias)
{
    int i = blockIdx.x * 256 + threadIdx.x;   // 0..1535
    float v;
    if (i < 256)      v = bq[i];
    else if (i < 512) v = bk[i - 256];
    else              v = bv[i - 512];
    bias[i] = v;
}

// ---------------- softmax over n; Sm stored [b][m][n] so rows are contiguous ----------
__global__ __launch_bounds__(256)
void softmax_row_kernel(const float* __restrict__ S, unsigned short* __restrict__ Attn)
{
    __shared__ float red[256];
    const size_t base = (size_t)blockIdx.x * 1024;
    const int t = threadIdx.x;
    float v[4];
    float mx = -3.4e38f;
    #pragma unroll
    for (int i = 0; i < 4; ++i) { v[i] = S[base + t + i * 256]; mx = fmaxf(mx, v[i]); }
    red[t] = mx; __syncthreads();
    for (int s = 128; s > 0; s >>= 1) {
        if (t < s) red[t] = fmaxf(red[t], red[t + s]);
        __syncthreads();
    }
    mx = red[0]; __syncthreads();
    float sum = 0.f;
    #pragma unroll
    for (int i = 0; i < 4; ++i) { v[i] = __expf(v[i] - mx); sum += v[i]; }
    red[t] = sum; __syncthreads();
    for (int s = 128; s > 0; s >>= 1) {
        if (t < s) red[t] += red[t + s];
        __syncthreads();
    }
    const float inv = 1.0f / red[0];
    #pragma unroll
    for (int i = 0; i < 4; ++i)
        Attn[base + t + i * 256] = f32_to_bf16_rne(v[i] * inv);
}

extern "C" void kernel_launch(void* const* d_in, const int* in_sizes, int n_in,
                              void* d_out, int out_size, void* d_ws, size_t ws_size,
                              hipStream_t stream)
{
    (void)in_sizes; (void)n_in; (void)out_size; (void)ws_size;
    const float* x  = (const float*)d_in[0];
    const float* Wq = (const float*)d_in[1];
    const float* bq = (const float*)d_in[2];
    const float* Wk = (const float*)d_in[3];
    const float* bk = (const float*)d_in[4];
    const float* Wv = (const float*)d_in[5];
    const float* bv = (const float*)d_in[6];
    float* out = (float*)d_out;

    const int B = 32, C = 1024, Cq = 256, N = 1024;
    const int Mqkv = C + 2 * Cq;   // 1536: rows [0,256)=Q, [256,512)=K, [512,1536)=V

    char* ws = (char*)d_ws;
    size_t off = 0;
    auto alloc = [&](size_t bytes) -> void* {
        void* p = ws + off;
        off = (off + bytes + 255) & ~(size_t)255;
        return p;
    };
    unsigned short* xT   = (unsigned short*)alloc((size_t)B * N * C * 2);         //  64 MB
    unsigned short* wb   = (unsigned short*)alloc((size_t)Mqkv * C * 2);          //   3 MB
    float*          bias = (float*)alloc((size_t)Mqkv * 4);
    unsigned short* qkt  = (unsigned short*)alloc((size_t)B * 2 * N * Cq * 2);    //  32 MB  [b][Qt|Kt][n][q]
    unsigned short* vbuf = (unsigned short*)alloc((size_t)B * C * N * 2);         //  64 MB  [b][c][n]
    float*          Sm   = (float*)alloc((size_t)B * N * N * 4);                  // 128 MB  [b][m][n]
    unsigned short* attn = (unsigned short*)alloc((size_t)B * N * N * 2);         //  64 MB  [b][m][n]

    // 1) transpose+convert x; convert fused weights; concat bias
    {
        dim3 grid(N / 32, C / 32, B);
        transpose_cvt_kernel<<<grid, 256, 0, stream>>>(x, xT, C, N);
    }
    cvt_f32_bf16_kernel<<<256, 256, 0, stream>>>(Wq, wb,                      (long long)Cq * C / 4);
    cvt_f32_bf16_kernel<<<256, 256, 0, stream>>>(Wk, wb + (size_t)Cq * C,     (long long)Cq * C / 4);
    cvt_f32_bf16_kernel<<<512, 256, 0, stream>>>(Wv, wb + (size_t)2 * Cq * C, (long long)C * C / 4);
    concat_bias_kernel<<<6, 256, 0, stream>>>(bq, bk, bv, bias);

    // 2) QKV projection: [1536x1024] x [1024x1024] per batch; Q/K stored transposed, V natural
    {
        dim3 grid(N / BN, Mqkv / BM, B);
        gemm_wmma_kernel<0><<<grid, NTHREADS, 0, stream>>>(
            wb, 0LL, C,                       // A: weights [M][K], shared across batch
            xT, (long long)N * C, C,          // B: xT[b] [n][c] -> direct [N][K] staging
            qkt, vbuf, 0LL, 0,
            bias, 0LL, C);
    }
    // 3) scores (transposed store): Sm[b][m][n] = sum_q Kt[b][m][q] * Qt[b][n][q]
    {
        dim3 grid(N / BN, N / BM, B);
        gemm_wmma_kernel<1><<<grid, NTHREADS, 0, stream>>>(
            qkt + (size_t)N * Cq, (long long)2 * N * Cq, Cq,   // A: Kt [m][q]
            qkt,                  (long long)2 * N * Cq, Cq,   // B: Qt [n][q]
            Sm, nullptr, (long long)N * N, N,
            nullptr, 0LL, Cq);
    }
    // 4) softmax over n (reference axis=1) == contiguous-row softmax of Sm; bf16 attn
    softmax_row_kernel<<<B * N, 256, 0, stream>>>(Sm, attn);

    // 5) out[b][c][m] = sigmoid( sum_n V[b][c][n] * attn[b][m][n] ) * x[b][c][m]
    {
        dim3 grid(N / BN, C / BM, B);
        gemm_wmma_kernel<2><<<grid, NTHREADS, 0, stream>>>(
            vbuf, (long long)C * N, N,        // A: V [c][n]
            attn, (long long)N * N, N,        // B: attn [m][n] -> direct [N][K] staging
            out, nullptr, (long long)C * N, N,
            x, (long long)C * N, N);          // K = N
    }
}
